// Image_MambaMultiscale_59270548685323
// MI455X (gfx1250) — matrix operand order repaired and checked by reading, not compiled, not run on hardware
//
#include <hip/hip_runtime.h>
#include <hip/hip_bf16.h>

// ---------------- problem dims ----------------
#define D_MODEL 64
#define D_INNER 128
#define D_STATE 16
#define D_CONV  4
#define NB      4
#define NH      24
#define NW      24
#define LSEQ    (NH*NW*21)      // 12096
#define BLTOT   ((long)NB*LSEQ) // 48384 rows
#define MTILES  (BLTOT/16)      // 3024
#define SCAN_CH 32              // scan chunk (steps staged per async round)

typedef __attribute__((ext_vector_type(16))) __bf16 v16bf;
typedef __attribute__((ext_vector_type(8)))  float  v8f;

// float -> bf16 bits (round to nearest even), stored as u16
static __device__ __forceinline__ unsigned short f2bf(float f) {
    unsigned int u = __float_as_uint(f);
    unsigned int r = (u + 0x7FFFu + ((u >> 16) & 1u)) >> 16;
    return (unsigned short)r;
}

union FragU { v16bf v; unsigned short e[16]; };

// A-matrix 16x32 bf16 fragment (wave32 layout): lane = m + 16*half
static __device__ __forceinline__ v16bf load_afrag(const unsigned short* __restrict__ row,
                                                   int half, int kbase) {
    FragU u;
#pragma unroll
    for (int i = 0; i < 8; ++i) {
        int kk = kbase + ((i < 4) ? (half * 8 + 2 * i) : (16 + half * 8 + 2 * (i - 4)));
        u.e[2 * i]     = row[kk];
        u.e[2 * i + 1] = row[kk + 1];
    }
    return u.v;
}

// B-matrix 32x16 bf16 fragment: lane = n + 16*half; wrow = weight row (length K) of out-channel n
static __device__ __forceinline__ v16bf load_bfrag(const unsigned short* __restrict__ wrow,
                                                   int half, int kbase) {
    FragU u;
#pragma unroll
    for (int e = 0; e < 16; ++e) u.e[e] = wrow[kbase + ((e < 8) ? (half * 8 + e) : (16 + half * 8 + (e - 8)))];
    return u.v;
}

static __device__ __forceinline__ v8f wmma_bf16(v16bf a, v16bf b, v8f c) {
    return __builtin_amdgcn_wmma_f32_16x16x32_bf16(false, a, false, b, (short)0, c, false, false);
}

// generic (shared) pointer -> LDS byte offset (low 32 bits of flat LDS-aperture address)
static __device__ __forceinline__ unsigned lds_off(const void* p) {
    return (unsigned)(unsigned long long)p;
}

// ---------------- k0: multiscale gather -> seq (b,l,c) bf16 ----------------
__global__ __launch_bounds__(256) void k_gather(const float* __restrict__ x2,
                                                const float* __restrict__ x3,
                                                const float* __restrict__ x4,
                                                unsigned short* __restrict__ seq) {
    long idx = (long)blockIdx.x * 256 + threadIdx.x;  // over B*L*64
    if (idx >= BLTOT * 64) return;
    int  c  = (int)(idx & 63);
    long bl = idx >> 6;
    int  b  = (int)(bl / LSEQ);
    int  l  = (int)(bl % LSEQ);
    int blk = l / 21, j = l % 21;
    int hb = blk / NW, wb = blk % NW;
    float v;
    if (j == 0) {
        v = x4[(((long)b * 64 + c) * NH + hb) * NW + wb];
    } else if (j < 5) {
        int p = j - 1, r = p >> 1, cc = p & 1;
        v = x3[(((long)b * 64 + c) * (NH * 2) + hb * 2 + r) * (NW * 2) + wb * 2 + cc];
    } else {
        int p = j - 5, r = p >> 2, cc = p & 3;
        v = x2[(((long)b * 64 + c) * (NH * 4) + hb * 4 + r) * (NW * 4) + wb * 4 + cc];
    }
    seq[idx] = f2bf(v);
}

// ---------------- k1: in_proj GEMM (64 -> 256), WMMA bf16 ----------------
__global__ __launch_bounds__(256) void k_inproj(const unsigned short* __restrict__ seq,
                                                const float* __restrict__ W,   // (256,64)
                                                float* __restrict__ xz) {      // (B*L,256)
    __shared__ unsigned short wl[256 * 64];
    int t = threadIdx.x;
    for (int i = t; i < 256 * 64; i += 256) wl[i] = f2bf(W[i]);
    __syncthreads();
    int wave = t >> 5, lane = t & 31;
    long row0 = ((long)blockIdx.x * 8 + wave) * 16;
    int m = lane & 15, half = lane >> 4;
    const unsigned short* arow = seq + (row0 + m) * 64;
    v16bf a0 = load_afrag(arow, half, 0);
    v16bf a1 = load_afrag(arow, half, 32);
#pragma unroll 4
    for (int nt = 0; nt < 16; ++nt) {
        const unsigned short* wrow = wl + (nt * 16 + m) * 64;
        v8f acc = {};
        acc = wmma_bf16(a0, load_bfrag(wrow, half, 0),  acc);
        acc = wmma_bf16(a1, load_bfrag(wrow, half, 32), acc);
#pragma unroll
        for (int r = 0; r < 8; ++r)
            xz[(row0 + r + half * 8) * 256 + nt * 16 + m] = acc[r];
    }
}

// ---------------- k2: depthwise causal conv(k=4) + SiLU ----------------
__global__ __launch_bounds__(256) void k_conv(const float* __restrict__ xz,
                                              const float* __restrict__ cw,   // (128,1,4)
                                              const float* __restrict__ cb,   // (128)
                                              float* __restrict__ xf,
                                              unsigned short* __restrict__ xb) {
    long idx = (long)blockIdx.x * 256 + threadIdx.x;  // over B*L*128
    if (idx >= BLTOT * 128) return;
    int  d  = (int)(idx & 127);
    long bl = idx >> 7;
    long b  = bl / LSEQ;
    long l  = bl % LSEQ;
    float acc = cb[d];
#pragma unroll
    for (int k = 0; k < 4; ++k) {
        long lt = l - 3 + k;
        if (lt >= 0) acc += cw[d * 4 + k] * xz[(b * LSEQ + lt) * 256 + d];
    }
    float s = acc / (1.f + __expf(-acc));
    xf[idx] = s;
    xb[idx] = f2bf(s);
}

// ---------------- kW: fold dt_proj into x_proj: W_dt = dt_proj_w @ x_proj_w[:4] ----------------
__global__ __launch_bounds__(256) void k_wdt(const float* __restrict__ xpw,   // (36,128)
                                             const float* __restrict__ dtw,   // (128,4)
                                             unsigned short* __restrict__ Wdt,// (128,128) bf16
                                             unsigned short* __restrict__ xpb)// (32,128) bf16
{
    int i = blockIdx.x * 256 + threadIdx.x;
    if (i < 128 * 128) {
        int d = i >> 7, c = i & 127;
        float a = 0.f;
#pragma unroll
        for (int r = 0; r < 4; ++r) a += dtw[d * 4 + r] * xpw[r * 128 + c];
        Wdt[i] = f2bf(a);
    }
    if (i < 32 * 128) xpb[i] = f2bf(xpw[4 * 128 + i]);
}

// ---------------- k3: fused dt (128->128, softplus) + B/C (128->32) WMMA ----------------
__global__ __launch_bounds__(256) void k_xproj(const unsigned short* __restrict__ xcb,
                                               const unsigned short* __restrict__ Wdt,
                                               const unsigned short* __restrict__ xpb,
                                               const float* __restrict__ dtb,    // (128)
                                               float* __restrict__ dtout,        // (B*L,128)
                                               float* __restrict__ BC) {         // (B*L,32)
    __shared__ unsigned short wd[128 * 128];
    __shared__ unsigned short wb[32 * 128];
    __shared__ float bias[128];
    int t = threadIdx.x;
    for (int i = t; i < 128 * 128; i += 256) wd[i] = Wdt[i];
    for (int i = t; i < 32 * 128;  i += 256) wb[i] = xpb[i];
    if (t < 128) bias[t] = dtb[t];
    __syncthreads();
    int wave = t >> 5, lane = t & 31;
    long row0 = ((long)blockIdx.x * 8 + wave) * 16;
    int m = lane & 15, half = lane >> 4;
    const unsigned short* arow = xcb + (row0 + m) * 128;
    v16bf af[4];
#pragma unroll
    for (int ks = 0; ks < 4; ++ks) af[ks] = load_afrag(arow, half, ks * 32);
#pragma unroll 2
    for (int nt = 0; nt < 8; ++nt) {
        const unsigned short* wrow = wd + (nt * 16 + m) * 128;
        v8f acc = {};
#pragma unroll
        for (int ks = 0; ks < 4; ++ks) acc = wmma_bf16(af[ks], load_bfrag(wrow, half, ks * 32), acc);
#pragma unroll
        for (int r = 0; r < 8; ++r) {
            float v = acc[r] + bias[nt * 16 + m];
            v = (v > 20.f) ? v : __logf(1.f + __expf(v));   // softplus
            dtout[(row0 + r + half * 8) * 128 + nt * 16 + m] = v;
        }
    }
#pragma unroll
    for (int nt = 0; nt < 2; ++nt) {
        const unsigned short* wrow = wb + (nt * 16 + m) * 128;
        v8f acc = {};
#pragma unroll
        for (int ks = 0; ks < 4; ++ks) acc = wmma_bf16(af[ks], load_bfrag(wrow, half, ks * 32), acc);
#pragma unroll
        for (int r = 0; r < 8; ++r)
            BC[(row0 + r + half * 8) * 32 + nt * 16 + m] = acc[r];
    }
}

// ---------------- k4: selective scan with async global->LDS double buffering ----------------
// grid = B*32 blocks; block = 64 thr = 2 waves; wave handles 2 adjacent channels
// (lane = dl*16 + n). Per chunk of SCAN_CH steps a wave issues 10 async-to-LDS ops
// (1x b64 dt, 1x b64 x, 8x b128 B/C) tracked by ASYNCcnt, consuming the previous
// chunk from LDS while the next streams in.
__global__ __launch_bounds__(64) void k_scan(const float* __restrict__ dt,
                                             const float* __restrict__ xf,
                                             const float* __restrict__ BCg,
                                             const float* __restrict__ A_log,
                                             float* __restrict__ y) {
    __shared__ __align__(16) float sdt[2][2][SCAN_CH * 2];
    __shared__ __align__(16) float sx [2][2][SCAN_CH * 2];
    __shared__ __align__(16) float sbc[2][2][SCAN_CH * 32];

    int b    = blockIdx.x >> 5;
    int dgrp = blockIdx.x & 31;
    int wave = threadIdx.x >> 5, lane = threadIdx.x & 31;
    int n = lane & 15, dl = lane >> 4;
    int d0 = dgrp * 4 + wave * 2;         // even -> 8B-aligned channel pair
    int d  = d0 + dl;

    float A = -__expf(A_log[d * 16 + n]);
    float h = 0.f;

    const unsigned long long dtbase = (unsigned long long)(const void*)(dt  + ((long)b * LSEQ) * 128 + d0);
    const unsigned long long xbase  = (unsigned long long)(const void*)(xf  + ((long)b * LSEQ) * 128 + d0);
    const unsigned long long bcbase = (unsigned long long)(const void*)(BCg + ((long)b * LSEQ) * 32);
    float* yp = y + (long)b * LSEQ * 128 + d;

    const unsigned vo_dtx = (unsigned)lane * (128u * 4u);                       // lane -> step lane
    const unsigned vo_bc  = (unsigned)(lane >> 3) * 128u + (unsigned)(lane & 7) * 16u;

    const int NCHK = LSEQ / SCAN_CH;      // 378

    auto issue = [&](int c, int buf) {
        unsigned long long row = (unsigned long long)c * SCAN_CH;
        unsigned long long dtb = dtbase + row * 128ull * 4ull;
        unsigned long long xb  = xbase  + row * 128ull * 4ull;
        unsigned long long bcb = bcbase + row * 32ull * 4ull;
        unsigned ldt = lds_off(&sdt[wave][buf][0]) + (unsigned)lane * 8u;
        unsigned lx  = lds_off(&sx [wave][buf][0]) + (unsigned)lane * 8u;
        asm volatile("global_load_async_to_lds_b64 %0, %1, %2"
                     :: "v"(ldt), "v"(vo_dtx), "s"(dtb) : "memory");
        asm volatile("global_load_async_to_lds_b64 %0, %1, %2"
                     :: "v"(lx), "v"(vo_dtx), "s"(xb) : "memory");
        unsigned lbc0 = lds_off(&sbc[wave][buf][0]);
#pragma unroll
        for (int j = 0; j < 8; ++j) {
            unsigned rel = (unsigned)j * 512u + vo_bc;   // (j*4 + lane/8)*128 + (lane&7)*16 bytes
            unsigned lb  = lbc0 + rel;
            asm volatile("global_load_async_to_lds_b128 %0, %1, %2"
                         :: "v"(lb), "v"(rel), "s"(bcb) : "memory");
        }
    };

    issue(0, 0);
    for (int c = 0; c < NCHK; ++c) {
        int buf = c & 1;
        if (c + 1 < NCHK) {
            issue(c + 1, buf ^ 1);
            // 10 ops outstanding for next chunk; in-order completion => prev chunk landed
            asm volatile("s_wait_asynccnt 10" ::: "memory");
        } else {
            asm volatile("s_wait_asynccnt 0" ::: "memory");
        }
        long lb = (long)c * SCAN_CH;
#pragma unroll 4
        for (int i = 0; i < SCAN_CH; ++i) {
            float dtv = sdt[wave][buf][i * 2 + dl];
            float xv  = sx [wave][buf][i * 2 + dl];
            float Bv  = sbc[wave][buf][i * 32 + n];
            float Cv  = sbc[wave][buf][i * 32 + 16 + n];
            float a = __expf(dtv * A);
            h = fmaf(a, h, dtv * xv * Bv);
            float yv = h * Cv;                        // reduce over 16 states
            yv += __shfl_xor(yv, 1, 16);
            yv += __shfl_xor(yv, 2, 16);
            yv += __shfl_xor(yv, 4, 16);
            yv += __shfl_xor(yv, 8, 16);
            if (n == 0) yp[(lb + i) * 128] = yv;
        }
    }
}

// ---------------- k5: gate (y + x*D)*silu(z) then out_proj (128->64) WMMA ----------------
__global__ __launch_bounds__(256) void k_outproj(const float* __restrict__ ysc,
                                                 const float* __restrict__ xf,
                                                 const float* __restrict__ xz,
                                                 const float* __restrict__ Dskip,
                                                 const float* __restrict__ Wout, // (64,128)
                                                 float* __restrict__ outseq) {   // (B*L,64)
    __shared__ unsigned short wl[64 * 128];
    __shared__ unsigned short gl[8][16 * 128];
    int t = threadIdx.x;
    for (int i = t; i < 64 * 128; i += 256) wl[i] = f2bf(Wout[i]);
    int wave = t >> 5, lane = t & 31;
    long row0 = ((long)blockIdx.x * 8 + wave) * 16;
    for (int i = lane; i < 16 * 128; i += 32) {
        int r = i >> 7, c = i & 127;
        long row = row0 + r;
        float yv = ysc[row * 128 + c];
        float xv = xf[row * 128 + c];
        float zv = xz[row * 256 + 128 + c];
        float g = (yv + xv * Dskip[c]) * (zv / (1.f + __expf(-zv)));
        gl[wave][i] = f2bf(g);
    }
    __syncthreads();
    int m = lane & 15, half = lane >> 4;
    const unsigned short* arow = &gl[wave][m * 128];
    v16bf af[4];
#pragma unroll
    for (int ks = 0; ks < 4; ++ks) af[ks] = load_afrag(arow, half, ks * 32);
#pragma unroll
    for (int nt = 0; nt < 4; ++nt) {
        const unsigned short* wrow = wl + (nt * 16 + m) * 128;
        v8f acc = {};
#pragma unroll
        for (int ks = 0; ks < 4; ++ks) acc = wmma_bf16(af[ks], load_bfrag(wrow, half, ks * 32), acc);
#pragma unroll
        for (int r = 0; r < 8; ++r)
            outseq[(row0 + r + half * 8) * 64 + nt * 16 + m] = acc[r];
    }
}

// ---------------- k6: scatter back to (B,C,96,96) + GroupNorm(32 groups of 2 ch) ----------------
__global__ __launch_bounds__(256) void k_gn(const float* __restrict__ outseq,
                                            const float* __restrict__ gnw,
                                            const float* __restrict__ gnb,
                                            float* __restrict__ out) {
    int b = blockIdx.x >> 5, g = blockIdx.x & 31;
    int c0 = g * 2, t = threadIdx.x;
    float s = 0.f, q = 0.f;
    for (int i = t; i < 2 * 9216; i += 256) {
        int c = c0 + i / 9216;
        int p = i % 9216;
        int h = p / 96, w = p % 96;
        int hb = h >> 2, r = h & 3, wb = w >> 2, cc = w & 3;
        long l = (long)(hb * NW + wb) * 21 + 5 + r * 4 + cc;
        float v = outseq[((long)b * LSEQ + l) * 64 + c];
        s += v; q += v * v;
    }
    __shared__ float rs[256], rq[256];
    rs[t] = s; rq[t] = q;
    __syncthreads();
    for (int o = 128; o > 0; o >>= 1) {
        if (t < o) { rs[t] += rs[t + o]; rq[t] += rq[t + o]; }
        __syncthreads();
    }
    float mean = rs[0] * (1.f / 18432.f);
    float var  = rq[0] * (1.f / 18432.f) - mean * mean;
    float rstd = rsqrtf(var + 1e-5f);
    for (int i = t; i < 2 * 9216; i += 256) {
        int c = c0 + i / 9216;
        int p = i % 9216;
        int h = p / 96, w = p % 96;
        int hb = h >> 2, r = h & 3, wb = w >> 2, cc = w & 3;
        long l = (long)(hb * NW + wb) * 21 + 5 + r * 4 + cc;
        float v = outseq[((long)b * LSEQ + l) * 64 + c];
        out[(((long)b * 64 + c) * 96 + h) * 96 + w] = (v - mean) * rstd * gnw[c] + gnb[c];
    }
}

// ---------------- host ----------------
extern "C" void kernel_launch(void* const* d_in, const int* in_sizes, int n_in,
                              void* d_out, int out_size, void* d_ws, size_t ws_size,
                              hipStream_t stream) {
    const float* x2    = (const float*)d_in[0];
    const float* x3    = (const float*)d_in[1];
    const float* x4    = (const float*)d_in[2];
    const float* ipw   = (const float*)d_in[3];
    const float* cw    = (const float*)d_in[4];
    const float* cb    = (const float*)d_in[5];
    const float* xpw   = (const float*)d_in[6];
    const float* dtw   = (const float*)d_in[7];
    const float* dtb   = (const float*)d_in[8];
    const float* A_log = (const float*)d_in[9];
    const float* Dsk   = (const float*)d_in[10];
    const float* opw   = (const float*)d_in[11];
    const float* gnw   = (const float*)d_in[12];
    const float* gnb   = (const float*)d_in[13];

    char* p = (char*)d_ws;
    auto alloc = [&](size_t bytes) { void* r = (void*)p; p += (bytes + 255) & ~(size_t)255; return r; };
    const long BL = BLTOT;
    unsigned short* seqb = (unsigned short*)alloc(BL * 64 * 2);
    float*          xz   = (float*)         alloc(BL * 256 * 4);
    float*          xcf  = (float*)         alloc(BL * 128 * 4);
    unsigned short* xcb  = (unsigned short*)alloc(BL * 128 * 2);
    float*          dtbf = (float*)         alloc(BL * 128 * 4);
    float*          BC   = (float*)         alloc(BL * 32 * 4);
    float*          ysc  = (float*)         alloc(BL * 128 * 4);
    float*          oseq = (float*)         alloc(BL * 64 * 4);
    unsigned short* Wdt  = (unsigned short*)alloc(128 * 128 * 2);
    unsigned short* xpb  = (unsigned short*)alloc(32 * 128 * 2);

    k_gather <<<(unsigned)((BL * 64 + 255) / 256), 256, 0, stream>>>(x2, x3, x4, seqb);
    k_inproj <<<(unsigned)(MTILES / 8), 256, 0, stream>>>(seqb, ipw, xz);
    k_conv   <<<(unsigned)((BL * 128 + 255) / 256), 256, 0, stream>>>(xz, cw, cb, xcf, xcb);
    k_wdt    <<<64, 256, 0, stream>>>(xpw, dtw, Wdt, xpb);
    k_xproj  <<<(unsigned)(MTILES / 8), 256, 0, stream>>>(xcb, Wdt, xpb, dtb, dtbf, BC);
    k_scan   <<<NB * 32, 64, 0, stream>>>(dtbf, xcf, BC, A_log, ysc);
    k_outproj<<<(unsigned)(MTILES / 8), 256, 0, stream>>>(ysc, xcf, xz, Dsk, opw, oseq);
    k_gn     <<<NB * 32, 256, 0, stream>>>(oseq, gnw, gnb, (float*)d_out);
}
